// CrossAttention_27376121545100
// MI455X (gfx1250) — matrix operand order repaired and checked by reading, not compiled, not run on hardware
//
#include <hip/hip_runtime.h>
#include <hip/hip_bf16.h>
#include <stdint.h>

// ---------------------------------------------------------------------------
// CDNA5 / gfx1250 cross-attention pipeline, f16 WMMA (f32 accumulate),
// TDM (tensor_load_to_lds) staging, double-buffered GEMM main loop.
// ---------------------------------------------------------------------------

typedef __attribute__((ext_vector_type(16))) _Float16 v16h;
typedef __attribute__((ext_vector_type(8)))  float    v8f;
typedef __attribute__((ext_vector_type(4)))  unsigned v4u;
typedef __attribute__((ext_vector_type(8)))  int      v8i;
typedef __attribute__((ext_vector_type(4)))  int      v4i;

#define HEADS    20
#define DIM_HEAD 64
#define QDIM     1280
#define CDIM     2048
#define INNER    1280
#define BATCH    16
#define NQ       4096
#define NK       77
#define NKP      80       // padded key length (5 x 16)
#define ATT_SCALE 0.125f  // 64^-0.5

// ---- workspace layout (units: _Float16 elements) --------------------------
static constexpr size_t WQ_OFF = 0;
static constexpr size_t WK_OFF = WQ_OFF + (size_t)INNER * QDIM;
static constexpr size_t WV_OFF = WK_OFF + (size_t)INNER * CDIM;
static constexpr size_t WO_OFF = WV_OFF + (size_t)INNER * CDIM;
static constexpr size_t Q_OFF  = WO_OFF + (size_t)QDIM * INNER;
static constexpr size_t K_OFF  = Q_OFF  + (size_t)BATCH * NQ  * INNER;
static constexpr size_t V_OFF  = K_OFF  + (size_t)BATCH * NKP * INNER;
static constexpr size_t A_OFF  = V_OFF  + (size_t)BATCH * NKP * INNER;

// ---------------------------------------------------------------------------
// TDM: 2-D tile load (global f16 -> LDS) per CDNA5 ISA.
// data_size = 2 bytes. pad_interval/pad_amount produce the padded LDS stride.
//   pad_interval code: 0=2,1=4,2=8,3=16,4=32,... DWORDs between pads
//   pad_amount   code: n -> (n+1) DWORDs of pad
// ---------------------------------------------------------------------------
__device__ __forceinline__ void tdm_load_2d(const _Float16* gsrc,
                                            unsigned lds_byte_off,
                                            unsigned rows, unsigned row_elems,
                                            unsigned long long stride_elems,
                                            int pad_interval, int pad_amount) {
  unsigned long long ga = (unsigned long long)(uintptr_t)gsrc;
  v4u g0;
  g0[0] = 1u;                                    // count=1, user mode
  g0[1] = lds_byte_off;                          // lds_addr
  g0[2] = (unsigned)(ga & 0xffffffffu);          // global_addr[31:0]
  g0[3] = (unsigned)((ga >> 32) & 0x01ffffffu)   // global_addr[56:32]
          | (2u << 30);                          // type = 2 ("image")
  v8i g1;
  g1[0] = (1 << 16)                              // data_size = 1 -> 2 bytes
        | (1 << 20)                              // pad_enable
        | ((pad_interval & 7) << 22)
        | ((pad_amount & 127) << 25);
  unsigned td0 = row_elems;                      // tensor_dim0 (tile in-bounds)
  unsigned td1 = rows;                           // tensor_dim1
  g1[1] = (int)((td0 & 0xffffu) << 16);          // bits[63:48] = td0 lo
  g1[2] = (int)(((td0 >> 16) & 0xffffu) | ((td1 & 0xffffu) << 16));
  g1[3] = (int)(((td1 >> 16) & 0xffffu) | ((row_elems & 0xffffu) << 16)); // tile_dim0
  g1[4] = (int)(rows & 0xffffu);                 // tile_dim1 (tile_dim2 = 0)
  g1[5] = (int)(stride_elems & 0xffffffffu);     // tensor_dim0_stride lo
  g1[6] = (int)((stride_elems >> 32) & 0xffffu); // stride hi; dim1_stride = 0
  g1[7] = 0;
  v4i z4 = {0, 0, 0, 0};
#if defined(__clang_major__) && (__clang_major__ >= 23)
  v8i z8 = {0, 0, 0, 0, 0, 0, 0, 0};
  __builtin_amdgcn_tensor_load_to_lds(g0, g1, z4, z4, z8, 0);
#else
  __builtin_amdgcn_tensor_load_to_lds(g0, g1, z4, z4, 0);
#endif
}

// ---------------------------------------------------------------------------
// WMMA helpers — wave32 fragments, A and B in contiguous k order (k = 16*(lane/16) + s)
// ---------------------------------------------------------------------------
__device__ __forceinline__ v8f wmma_f16(v16h a, v16h b, v8f c) {
  return __builtin_amdgcn_wmma_f32_16x16x32_f16(false, a, false, b, (short)0, c,
                                                false, false);
}

// A-matrix 16x32 f16 fragment from LDS tile laid out [m][k], row stride ld.
__device__ __forceinline__ v16h load_a_frag(const _Float16* base, int ld,
                                            int m0, int k0) {
  int lane = threadIdx.x & 31;
  int hf   = lane >> 4;
  int m    = m0 + (lane & 15);
  union { v16h h; unsigned u[8]; } f;
  const _Float16* row = base + m * ld + k0;
#pragma unroll
  for (int p = 0; p < 8; ++p) {
    int kk = hf * 16 + 2 * p;
    f.u[p] = *(const unsigned*)(row + kk);
  }
  return f.h;
}

// B-matrix 32x16 f16 fragment from LDS tile laid out [n][k], row stride ld.
__device__ __forceinline__ v16h load_b_frag(const _Float16* base, int ld,
                                            int n0, int k0) {
  int lane = threadIdx.x & 31;
  int hf   = lane >> 4;
  int n    = n0 + (lane & 15);
  union { v16h h; unsigned u[8]; } f;
  const _Float16* row = base + n * ld + k0 + hf * 16;
#pragma unroll
  for (int p = 0; p < 8; ++p) f.u[p] = *(const unsigned*)(row + 2 * p);
  return f.h;
}

// B-matrix fragment from LDS tile laid out [k][n] (e.g. V tile [nk][d]).
__device__ __forceinline__ v16h load_b_frag_t(const _Float16* base, int ld,
                                              int n0, int k0) {
  int lane = threadIdx.x & 31;
  int hf   = lane >> 4;
  int n    = n0 + (lane & 15);
  union { v16h h; _Float16 e[16]; } f;
  const _Float16* col = base + (k0 + hf * 16) * ld + n;
#pragma unroll
  for (int e = 0; e < 16; ++e) f.e[e] = col[e * ld];
  return f.h;
}

// ---------------------------------------------------------------------------
// Kernel 1: weight-norm all four matrices -> f16 (row-major [out][in])
// ---------------------------------------------------------------------------
__global__ __launch_bounds__(256) void wnorm_kernel(
    const float* __restrict__ vq, const float* __restrict__ gq,
    const float* __restrict__ vk, const float* __restrict__ gk,
    const float* __restrict__ vv, const float* __restrict__ gv,
    const float* __restrict__ vo, const float* __restrict__ go,
    _Float16* __restrict__ ws) {
  int row = blockIdx.x;
  int mat = blockIdx.y;
  const float* V; const float* G; _Float16* W; int in_dim;
  if      (mat == 0) { V = vq; G = gq; W = ws + WQ_OFF; in_dim = QDIM; }
  else if (mat == 1) { V = vk; G = gk; W = ws + WK_OFF; in_dim = CDIM; }
  else if (mat == 2) { V = vv; G = gv; W = ws + WV_OFF; in_dim = CDIM; }
  else               { V = vo; G = go; W = ws + WO_OFF; in_dim = INNER; }

  int tid = threadIdx.x, lane = tid & 31, wave = tid >> 5;
  const float* src = V + (size_t)row * in_dim;

  float ss = 0.f;
  for (int i = tid; i < in_dim; i += 256) { float x = src[i]; ss += x * x; }
#pragma unroll
  for (int d = 1; d < 32; d <<= 1) ss += __shfl_xor(ss, d, 32);

  __shared__ float red[8];
  __shared__ float stot;
  if (lane == 0) red[wave] = ss;
  __syncthreads();
  if (tid == 0) {
    float t = 0.f;
#pragma unroll
    for (int i = 0; i < 8; ++i) t += red[i];
    stot = t;
  }
  __syncthreads();

  float scale = G[row] * rsqrtf(stot);
  _Float16* dst = W + (size_t)row * in_dim;
  for (int i = tid; i < in_dim; i += 256) dst[i] = (_Float16)(src[i] * scale);
}

// ---------------------------------------------------------------------------
// Kernel 2: zero the K/V pad rows (nk = 77..79 per batch)
// ---------------------------------------------------------------------------
__global__ __launch_bounds__(256) void pad_kv_kernel(_Float16* __restrict__ ws) {
  int idx = blockIdx.x * 256 + threadIdx.x;
  const int total = BATCH * (NKP - NK) * INNER;
  if (idx >= total) return;
  int b   = idx / ((NKP - NK) * INNER);
  int rem = idx % ((NKP - NK) * INNER);
  int r   = NK + rem / INNER;
  int c   = rem % INNER;
  size_t off = ((size_t)(b * NKP + r)) * INNER + c;
  ws[K_OFF + off] = (_Float16)0.f;
  ws[V_OFF + off] = (_Float16)0.f;
}

// ---------------------------------------------------------------------------
// Kernel 3: tiled WMMA GEMM:  C[M,N] = A[M,K] * W[N,K]^T  (+bias)
// BM=64, BN=256, BK=32; 8 waves in 2(M) x 4(N); wave tile 32x64 -> 8 WMMA.
// Double-buffered pipeline: tile t+1 (A global loads + TDM B descriptor) is
// issued before the WMMAs of tile t; drained after; 1 barrier / iteration.
// ---------------------------------------------------------------------------
template <bool A_F32, bool OUT_F32, bool REMAP77>
__global__ __launch_bounds__(256) void gemm_wmma_kernel(
    const void* __restrict__ Av, const _Float16* __restrict__ Bw,
    void* __restrict__ Outv, const float* __restrict__ bias,
    int M, int N, int K) {
  constexpr int BM = 64, BN = 256, BK = 32;
  constexpr int LDA = BK + 8;   // 40 halves (TDM pad: 16 DW + 4 DW)
  constexpr int LDB = BK + 8;
  __shared__ alignas(16) _Float16 a_lds[2][BM * LDA];
  __shared__ alignas(16) _Float16 b_lds[2][BN * LDB];

  const int tid  = threadIdx.x;
  const int wave = tid >> 5;
  const int wm   = wave >> 2;   // 0..1
  const int wn   = wave & 3;    // 0..3
  const int m_blk = blockIdx.y * BM;
  const int n_blk = blockIdx.x * BN;

  const float*    Af = (const float*)Av;
  const _Float16* Ah = (const _Float16*)Av;

  v8f acc[2][4];
#pragma unroll
  for (int i = 0; i < 2; ++i)
#pragma unroll
    for (int j = 0; j < 4; ++j)
#pragma unroll
      for (int e = 0; e < 8; ++e) acc[i][j][e] = 0.f;

  const unsigned b_off0 = (unsigned)(uintptr_t)&b_lds[0][0];
  const unsigned b_off1 = (unsigned)(uintptr_t)&b_lds[1][0];

  const int arow = tid >> 2, aseg = tid & 3;   // A staging: 4 thr/row, 8 elems
  const int agm  = m_blk + arow;

  float areg[8];                               // A_F32 fetch registers
  uint4 aregh;                                 // f16 fetch register

  // ---- fetch A tile (global -> regs) --------------------------------------
  auto fetchA = [&](int k0) {
    if (A_F32) {
      if (agm < M) {
        const float* src = Af + (size_t)agm * K + k0 + aseg * 8;
        float4 f0 = *(const float4*)(src);
        float4 f1 = *(const float4*)(src + 4);
        areg[0] = f0.x; areg[1] = f0.y; areg[2] = f0.z; areg[3] = f0.w;
        areg[4] = f1.x; areg[5] = f1.y; areg[6] = f1.z; areg[7] = f1.w;
        if (k0 + BK < K) __builtin_prefetch(src + BK, 0, 1);
      } else {
#pragma unroll
        for (int p = 0; p < 8; ++p) areg[p] = 0.f;
      }
    } else {
      if (agm < M) {
        const _Float16* src = Ah + (size_t)agm * K + k0 + aseg * 8;
        aregh = *(const uint4*)src;
        if (k0 + BK < K) __builtin_prefetch(src + BK, 0, 1);
      } else {
        uint4 z = {0u, 0u, 0u, 0u};
        aregh = z;
      }
    }
  };
  // ---- drain A tile (regs -> LDS, with f32->f16 convert) ------------------
  auto storeA = [&](int buf) {
    _Float16* dst = &a_lds[buf][arow * LDA + aseg * 8];
    if (A_F32) {
#pragma unroll
      for (int p = 0; p < 8; ++p) dst[p] = (_Float16)areg[p];
    } else {
      *(uint4*)dst = aregh;
    }
  };

  const int nkt = K / BK;

  // ---- prologue: tile 0 ---------------------------------------------------
  fetchA(0);
  storeA(0);
  if (wave == 0) {
    tdm_load_2d(Bw + (size_t)n_blk * K, b_off0, BN, BK, (unsigned long long)K,
                /*pad_interval 16 DW*/ 3, /*pad 4 DW*/ 3);
    __builtin_amdgcn_s_wait_tensorcnt(0);
  }
  __syncthreads();

  // ---- pipelined main loop ------------------------------------------------
  for (int t = 0; t < nkt; ++t) {
    const int cur = t & 1, nxt = cur ^ 1;
    const bool has_next = (t + 1 < nkt);

    // issue tile t+1: A global loads into regs, TDM for next B buffer
    if (has_next) {
      fetchA((t + 1) * BK);
      if (wave == 0)
        tdm_load_2d(Bw + (size_t)n_blk * K + (size_t)(t + 1) * BK,
                    nxt ? b_off1 : b_off0, BN, BK, (unsigned long long)K, 3, 3);
    }

    // compute tile t: 8 WMMA per wave while next-tile loads are in flight
    v16h af[2], bf[4];
#pragma unroll
    for (int i = 0; i < 2; ++i)
      af[i] = load_a_frag(&a_lds[cur][0], LDA, wm * 32 + i * 16, 0);
#pragma unroll
    for (int j = 0; j < 4; ++j)
      bf[j] = load_b_frag(&b_lds[cur][0], LDB, wn * 64 + j * 16, 0);
#pragma unroll
    for (int i = 0; i < 2; ++i)
#pragma unroll
      for (int j = 0; j < 4; ++j)
        acc[i][j] = wmma_f16(af[i], bf[j], acc[i][j]);

    // drain tile t+1 into the alternate buffers
    if (has_next) {
      storeA(nxt);
      if (wave == 0) __builtin_amdgcn_s_wait_tensorcnt(0);
    }
    __syncthreads();
  }

  // ---- epilogue -----------------------------------------------------------
  const int lane = tid & 31, hf = lane >> 4, ln = lane & 15;
#pragma unroll
  for (int i = 0; i < 2; ++i) {
#pragma unroll
    for (int j = 0; j < 4; ++j) {
      int gn = n_blk + wn * 64 + j * 16 + ln;
#pragma unroll
      for (int r = 0; r < 8; ++r) {
        int gm = m_blk + wm * 32 + i * 16 + 8 * hf + r;
        if (gm < M) {
          float val = acc[i][j][r];
          if (OUT_F32) {
            ((float*)Outv)[(size_t)gm * N + gn] = val + bias[gn];
          } else {
            size_t orow = (size_t)gm;
            if (REMAP77) {
              int b = gm / NK, rr = gm % NK;
              orow = (size_t)(b * NKP + rr);
            }
            ((_Float16*)Outv)[orow * N + gn] = (_Float16)val;
          }
        }
      }
    }
  }
}

// ---------------------------------------------------------------------------
// Kernel 4: attention per (b, head, 64-row q tile); 4 waves, 16 q-rows each.
// Q/K/V tiles staged by TDM (one descriptor per wave 0/1/2).
// QK^T (WMMA) -> masked softmax (lane shuffles) -> P (via LDS) @ V (WMMA).
// ---------------------------------------------------------------------------
__global__ __launch_bounds__(128) void attn_kernel(
    const _Float16* __restrict__ qws, const _Float16* __restrict__ kws,
    const _Float16* __restrict__ vws, _Float16* __restrict__ aws) {
  constexpr int LD  = DIM_HEAD + 8;   // 72 halves (TDM pad: 32 DW + 4 DW)
  constexpr int PLD = 96;             // padded K for P@V (3 x 32)
  __shared__ alignas(16) _Float16 q_lds[64 * LD];
  __shared__ alignas(16) _Float16 k_lds[NKP * LD];
  __shared__ alignas(16) _Float16 v_lds[96 * LD];     // rows 80..95 zeroed
  __shared__ alignas(16) _Float16 p_lds[4 * 16 * PLD];

  const int tid = threadIdx.x;
  const int b = blockIdx.z, h = blockIdx.y, q0 = blockIdx.x * 64;
  const int wave = tid >> 5, lane = tid & 31, hf = lane >> 4, ln = lane & 15;

  // ---- TDM staging: wave0 -> Q, wave1 -> K, wave2 -> V --------------------
  if (wave == 0) {
    tdm_load_2d(qws + ((size_t)(b * NQ + q0)) * INNER + h * DIM_HEAD,
                (unsigned)(uintptr_t)&q_lds[0], 64, DIM_HEAD,
                (unsigned long long)INNER, /*32 DW*/ 4, /*4 DW*/ 3);
    __builtin_amdgcn_s_wait_tensorcnt(0);
  } else if (wave == 1) {
    tdm_load_2d(kws + ((size_t)(b * NKP)) * INNER + h * DIM_HEAD,
                (unsigned)(uintptr_t)&k_lds[0], NKP, DIM_HEAD,
                (unsigned long long)INNER, 4, 3);
    __builtin_amdgcn_s_wait_tensorcnt(0);
  } else if (wave == 2) {
    tdm_load_2d(vws + ((size_t)(b * NKP)) * INNER + h * DIM_HEAD,
                (unsigned)(uintptr_t)&v_lds[0], NKP, DIM_HEAD,
                (unsigned long long)INNER, 4, 3);
    __builtin_amdgcn_s_wait_tensorcnt(0);
  }
  // ---- zero V rows 80..95 (one uint4 per thread: 16 rows x 8 segs) --------
  {
    int row = 80 + (tid >> 3), seg = tid & 7;
    uint4 z = {0u, 0u, 0u, 0u};
    *(uint4*)(v_lds + row * LD + seg * 8) = z;
  }
  // ---- zero this wave's P pad columns (80..95) ----------------------------
  {
    _Float16* p = p_lds + wave * 16 * PLD;
    for (int t = lane; t < 16 * 8; t += 32) {
      int row = t >> 3, c = t & 7;
      *(unsigned*)(p + row * PLD + 80 + c * 2) = 0u;
    }
  }
  __syncthreads();

  // ---- S = Q K^T : 5 n-tiles x 2 k-steps = 10 WMMA ------------------------
  v16h aq0 = load_a_frag(q_lds, LD, wave * 16, 0);
  v16h aq1 = load_a_frag(q_lds, LD, wave * 16, 32);
  v8f s[5];
#pragma unroll
  for (int nt = 0; nt < 5; ++nt) {
#pragma unroll
    for (int e = 0; e < 8; ++e) s[nt][e] = 0.f;
    v16h bk0 = load_b_frag(k_lds, LD, nt * 16, 0);
    v16h bk1 = load_b_frag(k_lds, LD, nt * 16, 32);
    s[nt] = wmma_f16(aq0, bk0, s[nt]);
    s[nt] = wmma_f16(aq1, bk1, s[nt]);
  }

  // ---- scale + mask columns >= 77 -----------------------------------------
#pragma unroll
  for (int nt = 0; nt < 5; ++nt) {
    int nk = nt * 16 + ln;
    bool oob = (nk >= NK);
#pragma unroll
    for (int r = 0; r < 8; ++r)
      s[nt][r] = oob ? -3.0e38f : s[nt][r] * ATT_SCALE;
  }

  // ---- softmax over 80 cols: reduce across the 16-lane half ---------------
  float inv_sum[8];
#pragma unroll
  for (int r = 0; r < 8; ++r) {
    float mx = s[0][r];
#pragma unroll
    for (int nt = 1; nt < 5; ++nt) mx = fmaxf(mx, s[nt][r]);
#pragma unroll
    for (int d = 1; d < 16; d <<= 1) mx = fmaxf(mx, __shfl_xor(mx, d, 32));
    float sum = 0.f;
#pragma unroll
    for (int nt = 0; nt < 5; ++nt) {
      float e = __expf(s[nt][r] - mx);
      s[nt][r] = e;
      sum += e;
    }
#pragma unroll
    for (int d = 1; d < 16; d <<= 1) sum += __shfl_xor(sum, d, 32);
    inv_sum[r] = 1.f / sum;
  }

  // ---- P -> LDS (transpose from C-layout to A-layout via memory) ----------
  _Float16* p = p_lds + wave * 16 * PLD;
#pragma unroll
  for (int nt = 0; nt < 5; ++nt)
#pragma unroll
    for (int r = 0; r < 8; ++r) {
      int m = r + 8 * hf;
      p[m * PLD + nt * 16 + ln] = (_Float16)(s[nt][r] * inv_sum[r]);
    }
  __syncthreads();

  // ---- O = P V : 3 k-steps x 4 n-tiles = 12 WMMA --------------------------
  v8f o[4];
#pragma unroll
  for (int nt = 0; nt < 4; ++nt)
#pragma unroll
    for (int e = 0; e < 8; ++e) o[nt][e] = 0.f;
#pragma unroll
  for (int t = 0; t < 3; ++t) {
    v16h ap = load_a_frag(p, PLD, 0, t * 32);
#pragma unroll
    for (int nt = 0; nt < 4; ++nt) {
      v16h bv = load_b_frag_t(v_lds, LD, nt * 16, t * 32);
      o[nt] = wmma_f16(ap, bv, o[nt]);
    }
  }

  // ---- write attention output tile (f16) ----------------------------------
#pragma unroll
  for (int nt = 0; nt < 4; ++nt)
#pragma unroll
    for (int r = 0; r < 8; ++r) {
      int m = r + 8 * hf;
      size_t row = (size_t)(b * NQ + q0 + wave * 16 + m);
      aws[row * INNER + h * DIM_HEAD + nt * 16 + ln] = (_Float16)o[nt][r];
    }
}

// ---------------------------------------------------------------------------
// Host-side launch
// ---------------------------------------------------------------------------
extern "C" void kernel_launch(void* const* d_in, const int* in_sizes, int n_in,
                              void* d_out, int out_size, void* d_ws,
                              size_t ws_size, hipStream_t stream) {
  const float* x   = (const float*)d_in[0];
  const float* ctx = (const float*)d_in[1];
  const float* vq  = (const float*)d_in[2];
  const float* gq  = (const float*)d_in[3];
  const float* vk  = (const float*)d_in[4];
  const float* gk  = (const float*)d_in[5];
  const float* vv  = (const float*)d_in[6];
  const float* gv  = (const float*)d_in[7];
  const float* vo  = (const float*)d_in[8];
  const float* go  = (const float*)d_in[9];
  const float* bo  = (const float*)d_in[10];

  _Float16* ws  = (_Float16*)d_ws;
  float*    out = (float*)d_out;

  // 1) weight norm -> f16 weights
  wnorm_kernel<<<dim3(INNER, 4), 256, 0, stream>>>(vq, gq, vk, gk, vv, gv, vo,
                                                   go, ws);
  // 2) zero K/V pad rows
  {
    int total = BATCH * (NKP - NK) * INNER;
    pad_kv_kernel<<<(total + 255) / 256, 256, 0, stream>>>(ws);
  }
  // 3) Q projection: [65536,1280] = x * Wq^T  (f32 in, f16 out)
  gemm_wmma_kernel<true, false, false>
      <<<dim3(INNER / 256, (BATCH * NQ) / 64), 256, 0, stream>>>(
          x, ws + WQ_OFF, (void*)(ws + Q_OFF), nullptr, BATCH * NQ, INNER, QDIM);
  // 4) K projection: [1232,1280] = ctx * Wk^T (rows remapped 77 -> 80)
  gemm_wmma_kernel<true, false, true>
      <<<dim3(INNER / 256, (BATCH * NK + 63) / 64), 256, 0, stream>>>(
          ctx, ws + WK_OFF, (void*)(ws + K_OFF), nullptr, BATCH * NK, INNER,
          CDIM);
  // 5) V projection
  gemm_wmma_kernel<true, false, true>
      <<<dim3(INNER / 256, (BATCH * NK + 63) / 64), 256, 0, stream>>>(
          ctx, ws + WV_OFF, (void*)(ws + V_OFF), nullptr, BATCH * NK, INNER,
          CDIM);
  // 6) attention
  attn_kernel<<<dim3(NQ / 64, HEADS, BATCH), 128, 0, stream>>>(
      ws + Q_OFF, ws + K_OFF, ws + V_OFF, ws + A_OFF);
  // 7) output projection + bias -> f32 d_out
  gemm_wmma_kernel<false, true, false>
      <<<dim3(QDIM / 256, (BATCH * NQ) / 64), 256, 0, stream>>>(
          (const void*)(ws + A_OFF), ws + WO_OFF, (void*)out, bo, BATCH * NQ,
          QDIM, INNER);
}
